// MoEBase_42391327212229
// MI455X (gfx1250) — matrix operand order repaired and checked by reading, not compiled, not run on hardware
//
#include <hip/hip_runtime.h>
#include <hip/hip_bf16.h>

#define D_MODEL   1024
#define N_EXPERTS 8
#define HIDDEN    2048
#define T_TOKENS  4096
#define ROWS      16                  // token-tile (WMMA M)
#define NT        520                 // max 16-row tiles: 8192/16 + 8 pad tiles
#define BUCKET_CAP (NT * ROWS)

typedef __attribute__((ext_vector_type(16))) __bf16 v16bf;
typedef __attribute__((ext_vector_type(8)))  float  v8f;
typedef __attribute__((ext_vector_type(4)))  unsigned int v4u;
typedef __attribute__((ext_vector_type(8)))  int    v8i;
typedef __attribute__((ext_vector_type(4)))  int    v4i;

union AFrag { v16bf v; uint4 q[2]; };
union BFrag { v16bf v; unsigned int p[8]; };

#if __has_builtin(__builtin_amdgcn_tensor_load_to_lds) && __has_builtin(__builtin_amdgcn_s_wait_tensorcnt)
#define USE_TDM 1
#endif

__device__ __forceinline__ unsigned short f2bf(float f) {
    union { float f; unsigned u; } x; x.f = f;
    unsigned r = x.u + 0x7FFFu + ((x.u >> 16) & 1u);   // round-to-nearest-even
    return (unsigned short)(r >> 16);
}

// RNE-convert two floats to bf16 and pack (hi<<16)|lo with one v_perm_b32
__device__ __forceinline__ unsigned pack2bf(float lo, float hi) {
    union { float f; unsigned u; } a, b; a.f = lo; b.f = hi;
    unsigned ra = a.u + 0x7FFFu + ((a.u >> 16) & 1u);
    unsigned rb = b.u + 0x7FFFu + ((b.u >> 16) & 1u);
    return __builtin_amdgcn_perm(rb, ra, 0x07060302u); // {rb[31:16], ra[31:16]}
}

// ---------------- kernel 1: x fp32 -> bf16, zero output, zero counters -------
__global__ __launch_bounds__(256) void k_convert_zero(
    const float* __restrict__ x, unsigned short* __restrict__ xb,
    float* __restrict__ out, int* __restrict__ counts, int* __restrict__ cursors)
{
    int i = (blockIdx.x * 256 + threadIdx.x) * 4;      // covers T*D = 4M elems
    float4 v = *(const float4*)(x + i);
    uint2 p;
    p.x = pack2bf(v.x, v.y);
    p.y = pack2bf(v.z, v.w);
    *(uint2*)(xb + i) = p;
    float4 z = make_float4(0.f, 0.f, 0.f, 0.f);
    *(float4*)(out + i) = z;
    if (blockIdx.x == 0 && threadIdx.x < N_EXPERTS) {
        counts[threadIdx.x] = 0;
        cursors[threadIdx.x] = 0;
    }
}

// ---------------- kernel 2: router (logits -> softmax -> top2) ---------------
__global__ __launch_bounds__(256) void k_router(
    const float* __restrict__ x, const float* __restrict__ wr,
    int* __restrict__ top_idx, float* __restrict__ top_w, int* __restrict__ counts)
{
    int t = blockIdx.x * 256 + threadIdx.x;            // one thread per token
    const float4* xr = (const float4*)(x + (size_t)t * D_MODEL);
    float acc[N_EXPERTS];
    #pragma unroll
    for (int e = 0; e < N_EXPERTS; ++e) acc[e] = 0.f;
    for (int d4 = 0; d4 < D_MODEL / 4; ++d4) {
        float4 v = xr[d4];
        const float* w0 = wr + (size_t)d4 * 4 * N_EXPERTS;
        #pragma unroll
        for (int e = 0; e < N_EXPERTS; ++e)
            acc[e] += v.x * w0[e] + v.y * w0[N_EXPERTS + e]
                    + v.z * w0[2 * N_EXPERTS + e] + v.w * w0[3 * N_EXPERTS + e];
    }
    float m = acc[0];
    #pragma unroll
    for (int e = 1; e < N_EXPERTS; ++e) m = fmaxf(m, acc[e]);
    float p[N_EXPERTS], s = 0.f;
    #pragma unroll
    for (int e = 0; e < N_EXPERTS; ++e) { p[e] = __expf(acc[e] - m); s += p[e]; }
    float inv = __builtin_amdgcn_rcpf(s);
    float b1 = -1.f; int i1 = 0;
    #pragma unroll
    for (int e = 0; e < N_EXPERTS; ++e) if (p[e] > b1) { b1 = p[e]; i1 = e; }
    float b2 = -1.f; int i2 = 0;
    #pragma unroll
    for (int e = 0; e < N_EXPERTS; ++e) if (e != i1 && p[e] > b2) { b2 = p[e]; i2 = e; }
    top_idx[2 * t]     = i1;  top_w[2 * t]     = b1 * inv;
    top_idx[2 * t + 1] = i2;  top_w[2 * t + 1] = b2 * inv;
    atomicAdd(counts + i1, 1);
    atomicAdd(counts + i2, 1);
}

// ---------------- kernel 3: offsets (pad to 16) + bucket prefill -------------
__global__ void k_offsets(const int* __restrict__ counts, int* __restrict__ offsets,
                          int* __restrict__ btok, float* __restrict__ bw)
{
    int tid = threadIdx.x;
    if (tid == 0) {
        int cum = 0;
        #pragma unroll
        for (int e = 0; e < N_EXPERTS; ++e) {
            offsets[e] = cum;
            cum += (counts[e] + 15) & ~15;
        }
        offsets[N_EXPERTS] = cum;
    }
    for (int i = tid; i < BUCKET_CAP; i += blockDim.x) { btok[i] = 0; bw[i] = 0.f; }
}

// ---------------- kernel 4: scatter (token,k) into expert buckets ------------
__global__ __launch_bounds__(256) void k_scatter(
    const int* __restrict__ top_idx, const float* __restrict__ top_w,
    const int* __restrict__ offsets, int* __restrict__ cursors,
    int* __restrict__ btok, float* __restrict__ bw)
{
    int i = blockIdx.x * 256 + threadIdx.x;            // T*2 pairs
    if (i >= T_TOKENS * 2) return;
    int e = top_idx[i];
    int pos = atomicAdd(cursors + e, 1);
    int slot = offsets[e] + pos;
    btok[slot] = i >> 1;
    bw[slot] = top_w[i];
}

// ---------------- kernel 5: fused SwiGLU MoE tile (bf16 WMMA) ----------------
__global__ __launch_bounds__(256) void k_moe(
    const unsigned short* __restrict__ xb,
    const float* __restrict__ w1, const float* __restrict__ w3,
    const float* __restrict__ w2,
    const int* __restrict__ offsets, const int* __restrict__ btok,
    const float* __restrict__ bw, float* __restrict__ out)
{
    __shared__ unsigned short sx[ROWS * D_MODEL];      // 32 KB: gathered x tile (bf16)
    __shared__ unsigned short sh[ROWS * HIDDEN];       // 64 KB: h = silu(xw1)*xw3 (bf16)

    const int tile = blockIdx.x;
    const int tid  = threadIdx.x;
    const int wave = tid >> 5;
    const int lane = tid & 31;
    const int row0 = tile * ROWS;

    // expert for this tile (buckets padded to 16, so tiles never straddle experts)
    int e = 0;
    #pragma unroll
    for (int i = 0; i < N_EXPERTS - 1; ++i) e += (row0 >= offsets[i + 1]) ? 1 : 0;
    const float* w1e = w1 + (size_t)e * D_MODEL * HIDDEN;
    const float* w3e = w3 + (size_t)e * D_MODEL * HIDDEN;
    const float* w2e = w2 + (size_t)e * HIDDEN * D_MODEL;

    // ---- stage A: gather 16 token rows (bf16) into LDS --------------------
#ifdef USE_TDM
    if (wave == 0) {
        int tk[16];
        #pragma unroll
        for (int r = 0; r < 16; ++r) tk[r] = btok[row0 + r];
        unsigned ldsoff = (unsigned)(unsigned long long)&sx[0];
        unsigned long long ga = (unsigned long long)xb;
        v4u g0; v8i g1; v4i g2, g3;
        // D# group0: count=1, gather_mode=1 (16-bit idx), lds_addr, global_addr, type=2
        g0[0] = 0x80000001u;
        g0[1] = ldsoff;
        g0[2] = (unsigned)(ga & 0xffffffffu);
        g0[3] = (unsigned)((ga >> 32) & 0x01ffffffu) | (2u << 30);
        // D# group1: data_size=2B; tensor_dim0=1024; tensor_dim1=4096;
        //            tile_dim0=1024; tile_dim1=16 indices; dim0_stride=1024
        g1[0] = (int)(1u << 16);
        g1[1] = (int)((unsigned)D_MODEL << 16);
        g1[2] = (int)((unsigned)T_TOKENS << 16);
        g1[3] = (int)((unsigned)D_MODEL << 16);
        g1[4] = 16;
        g1[5] = D_MODEL;
        g1[6] = 0;
        g1[7] = 0;
        // D# groups2/3: 16 x 16-bit row (token) indices
        g2[0] = (tk[0]  & 0xffff) | (tk[1]  << 16);
        g2[1] = (tk[2]  & 0xffff) | (tk[3]  << 16);
        g2[2] = (tk[4]  & 0xffff) | (tk[5]  << 16);
        g2[3] = (tk[6]  & 0xffff) | (tk[7]  << 16);
        g3[0] = (tk[8]  & 0xffff) | (tk[9]  << 16);
        g3[1] = (tk[10] & 0xffff) | (tk[11] << 16);
        g3[2] = (tk[12] & 0xffff) | (tk[13] << 16);
        g3[3] = (tk[14] & 0xffff) | (tk[15] << 16);
        v8i gz = {0, 0, 0, 0, 0, 0, 0, 0};             // extra group (6-arg toolchain ABI)
        __builtin_amdgcn_tensor_load_to_lds(g0, g1, g2, g3, gz, 0);
        __builtin_amdgcn_s_wait_tensorcnt(0);
    }
#else
    {
        uint4* dst = (uint4*)sx;                       // 2048 uint4
        #pragma unroll
        for (int it = 0; it < 8; ++it) {
            int i = tid + it * 256;
            int r = i >> 7, c = i & 127;               // 128 uint4 per row
            int tok = btok[row0 + r];
            dst[i] = ((const uint4*)(xb + (size_t)tok * D_MODEL))[c];
        }
    }
#endif
    __syncthreads();

    const int m   = lane & 15;                         // A-frag row
    const int kb  = (lane < 16) ? 0 : 16;              // A-frag K sub-offset
    const int kbB = (lane < 16) ? 0 : 16;              // B-frag K sub-offset
    const int n   = lane & 15;                         // B/C/D column

    // ---- GEMM1: h[16 x 2048] = silu(x w1) * (x w3); wave owns 256 H-cols ---
    for (int t = 0; t < 16; ++t) {
        int n0 = wave * 256 + t * 16;
        v8f acc1 = {}; v8f acc3 = {};
        for (int k0 = 0; k0 < D_MODEL; k0 += 32) {
            AFrag a;
            a.q[0] = *(const uint4*)&sx[m * D_MODEL + k0 + kb];
            a.q[1] = *(const uint4*)&sx[m * D_MODEL + k0 + 8 + kb];
            const float* p1 = w1e + (size_t)(k0 + kbB) * HIDDEN + n0 + n;
            const float* p3 = w3e + (size_t)(k0 + kbB) * HIDDEN + n0 + n;
            __builtin_prefetch(p1 + (size_t)32 * HIDDEN, 0, 1);
            __builtin_prefetch(p3 + (size_t)32 * HIDDEN, 0, 1);
            BFrag b1, b3;
            #pragma unroll
            for (int j = 0; j < 8; ++j) {
                b1.p[j] = pack2bf(p1[(size_t)(2 * j) * HIDDEN], p1[(size_t)(2 * j + 1) * HIDDEN]);
                b3.p[j] = pack2bf(p3[(size_t)(2 * j) * HIDDEN], p3[(size_t)(2 * j + 1) * HIDDEN]);
            }
            acc1 = __builtin_amdgcn_wmma_f32_16x16x32_bf16(false, a.v, false, b1.v,
                                                           (short)0, acc1, false, false);
            acc3 = __builtin_amdgcn_wmma_f32_16x16x32_bf16(false, a.v, false, b3.v,
                                                           (short)0, acc3, false, false);
        }
        #pragma unroll
        for (int r = 0; r < 8; ++r) {
            float g   = acc1[r];
            float sig = __builtin_amdgcn_rcpf(1.f + __expf(-g));   // fast sigmoid
            float hv  = g * sig * acc3[r];
            int M = r + ((lane < 16) ? 0 : 8);
            sh[M * HIDDEN + n0 + n] = f2bf(hv);
        }
    }
    __syncthreads();

    // routing weights / token ids for this tile (per-lane copies, 8 rows each)
    int   tokr[8];
    float wvr[8];
    #pragma unroll
    for (int r = 0; r < 8; ++r) {
        int M = r + ((lane < 16) ? 0 : 8);
        tokr[r] = btok[row0 + M];
        wvr[r]  = bw[row0 + M];
    }

    // ---- GEMM2: y[16 x 1024] = h w2, scale, atomic combine -----------------
    for (int s = 0; s < 8; ++s) {
        int n0 = wave * 128 + s * 16;
        v8f acc = {};
        for (int k0 = 0; k0 < HIDDEN; k0 += 32) {
            AFrag a;
            a.q[0] = *(const uint4*)&sh[m * HIDDEN + k0 + kb];
            a.q[1] = *(const uint4*)&sh[m * HIDDEN + k0 + 8 + kb];
            const float* p2 = w2e + (size_t)(k0 + kbB) * D_MODEL + n0 + n;
            __builtin_prefetch(p2 + (size_t)32 * D_MODEL, 0, 1);
            BFrag b;
            #pragma unroll
            for (int j = 0; j < 8; ++j)
                b.p[j] = pack2bf(p2[(size_t)(2 * j) * D_MODEL], p2[(size_t)(2 * j + 1) * D_MODEL]);
            acc = __builtin_amdgcn_wmma_f32_16x16x32_bf16(false, a.v, false, b.v,
                                                          (short)0, acc, false, false);
        }
        #pragma unroll
        for (int r = 0; r < 8; ++r) {
            atomicAdd(out + (size_t)tokr[r] * D_MODEL + n0 + n, acc[r] * wvr[r]);
        }
    }
}

// ---------------------------------------------------------------------------
extern "C" void kernel_launch(void* const* d_in, const int* in_sizes, int n_in,
                              void* d_out, int out_size, void* d_ws, size_t ws_size,
                              hipStream_t stream) {
    const float* x  = (const float*)d_in[0];   // [2,2048,1024]
    const float* wr = (const float*)d_in[1];   // [1024,8]
    const float* w1 = (const float*)d_in[2];   // [8,1024,2048]
    const float* w3 = (const float*)d_in[3];   // [8,1024,2048]
    const float* w2 = (const float*)d_in[4];   // [8,2048,1024]
    float* out = (float*)d_out;                // [2,2048,1024] fp32

    char* ws = (char*)d_ws;
    size_t off = 0;
    unsigned short* xb = (unsigned short*)(ws + off); off += (size_t)T_TOKENS * D_MODEL * 2;
    int*   top_idx = (int*)(ws + off);   off += (size_t)T_TOKENS * 2 * 4;
    float* top_w   = (float*)(ws + off); off += (size_t)T_TOKENS * 2 * 4;
    int*   counts  = (int*)(ws + off);   off += 64;
    int*   offsets = (int*)(ws + off);   off += 64;
    int*   cursors = (int*)(ws + off);   off += 64;
    int*   btok    = (int*)(ws + off);   off += (size_t)BUCKET_CAP * 4;
    float* bwt     = (float*)(ws + off); off += (size_t)BUCKET_CAP * 4;

    k_convert_zero<<<(T_TOKENS * D_MODEL / 4) / 256, 256, 0, stream>>>(x, xb, out, counts, cursors);
    k_router<<<T_TOKENS / 256, 256, 0, stream>>>(x, wr, top_idx, top_w, counts);
    k_offsets<<<1, 256, 0, stream>>>(counts, offsets, btok, bwt);
    k_scatter<<<(T_TOKENS * 2) / 256, 256, 0, stream>>>(top_idx, top_w, offsets, cursors, btok, bwt);
    k_moe<<<NT, 256, 0, stream>>>(xb, w1, w3, w2, offsets, btok, bwt, out);
}
